// pointcnn_15616501088308
// MI455X (gfx1250) — matrix operand order repaired and checked by reading, not compiled, not run on hardware
//
#include <hip/hip_runtime.h>
#include <hip/hip_bf16.h>

#define BATCH 8
#define NPTS  4096
#define KNN   16
#define COUT  64

typedef _Float16 v8h  __attribute__((ext_vector_type(8)));
typedef _Float16 v16h __attribute__((ext_vector_type(16)));
typedef float    v8f  __attribute__((ext_vector_type(8)));

// Load 16 consecutive f32 and convert to a 16-half WMMA fragment register set.
__device__ __forceinline__ v16h cvt_frag16(const float* __restrict__ p) {
  v16h r;
#pragma unroll
  for (int i = 0; i < 16; ++i) r[i] = (_Float16)p[i];
  return r;
}

__device__ __forceinline__ v16h combine8(v8h lo, v8h hi) {
  v16h r;
#pragma unroll
  for (int i = 0; i < 8; ++i) { r[i] = lo[i]; r[i + 8] = hi[i]; }
  return r;
}

// --------------------------------------------------------------------------
// Kernel 1: exact kNN (16 nearest, excluding self). One thread per query.
// Candidates tiled through SoA LDS with precomputed norms:
//   argmin over p of |p|^2 - 2 q.p   (== argmin |q-p|^2)
// Inner loop: 3 FMA + compare. Top-16 kept as unrolled insertion list in regs.
// --------------------------------------------------------------------------
__global__ __launch_bounds__(256) void knn_kernel(const float* __restrict__ xyz,
                                                  int* __restrict__ idxOut) {
  const int tid = threadIdx.x;
  const int b   = blockIdx.x >> 4;                 // 16 blocks per batch
  const int q   = ((blockIdx.x & 15) << 8) + tid;  // query index in [0,4096)
  const float* base = xyz + (size_t)b * NPTS * 3;

  const float mqx = -2.0f * base[q * 3 + 0];
  const float mqy = -2.0f * base[q * 3 + 1];
  const float mqz = -2.0f * base[q * 3 + 2];

  float bd[KNN];
  int   bi[KNN];
#pragma unroll
  for (int i = 0; i < KNN; ++i) { bd[i] = 3.4e38f; bi[i] = 0; }

  __shared__ float tx[256], ty[256], tz[256], tn[256];

  for (int t0 = 0; t0 < NPTS; t0 += 256) {
    __syncthreads();
    const float px = base[(t0 + tid) * 3 + 0];
    const float py = base[(t0 + tid) * 3 + 1];
    const float pz = base[(t0 + tid) * 3 + 2];
    tx[tid] = px; ty[tid] = py; tz[tid] = pz;
    tn[tid] = fmaf(px, px, fmaf(py, py, pz * pz));
    __syncthreads();
#pragma unroll 4
    for (int j = 0; j < 256; ++j) {
      const int p = t0 + j;
      const float d = fmaf(mqx, tx[j], fmaf(mqy, ty[j], fmaf(mqz, tz[j], tn[j])));
      if (d < bd[KNN - 1] && p != q) {
#pragma unroll
        for (int s = KNN - 1; s >= 1; --s) {
          const bool cs = d < bd[s];
          const bool cp = d < bd[s - 1];
          if (cs) { bd[s] = cp ? bd[s - 1] : d; bi[s] = cp ? bi[s - 1] : p; }
        }
        if (d < bd[0]) { bd[0] = d; bi[0] = p; }
      }
    }
  }

  int* o = idxOut + ((size_t)(b * NPTS + q) << 4);
#pragma unroll
  for (int i = 0; i < KNN; ++i) o[i] = bi[i];
}

// --------------------------------------------------------------------------
// Kernel 2: grouped MLP with WMMA. One wave32 per point (grid-strided).
//  * launch_bounds(256, 4): >=4 waves/SIMD -> VGPR cap <=256, so the
//    allocator cannot slip into the V256+ bank (no s_set_vgpr_msb traffic).
//  * W2 fragments (8 x 8 VGPRs) register-resident -> layer-2 WMMAs wait-free.
//  * W3 staged in LDS as f16; its B fragments loaded per tile (2x ds_load_b128)
//    with the previous tile's max-reduction hiding the LDS latency.
//  * Layer 1 (3->64) computes exactly the channels each lane's A fragment
//    needs, so layer-2 A fragments are built directly in registers.
//  * Only the layer2->layer3 activation transpose goes through LDS.
// --------------------------------------------------------------------------
__global__ __launch_bounds__(256, 4) void mlp_kernel(const float* __restrict__ xyz,
                                                     const float* __restrict__ W1,
                                                     const float* __restrict__ W2,
                                                     const float* __restrict__ W3,
                                                     const int* __restrict__ idx,
                                                     float* __restrict__ out) {
  __shared__ float W1s[COUT * 3];
  __shared__ __attribute__((aligned(16))) _Float16 W3h[COUT * COUT];
  __shared__ __attribute__((aligned(16))) _Float16 hbuf[8 * KNN * COUT];

  const int tid = threadIdx.x;
  for (int i = tid; i < COUT * 3; i += 256) W1s[i] = W1[i];
  for (int i = tid; i < COUT * COUT; i += 256) W3h[i] = (_Float16)W3[i];
  __syncthreads();

  const int lane = tid & 31;
  const int wv   = tid >> 5;
  const int hi   = (lane >> 4) & 1;   // which 16-lane half
  const int m16  = lane & 15;
  _Float16* H = hbuf + wv * (KNN * COUT);

  // ---- preload W2 fragments (loop-invariant) into registers --------------
  // B fragment (32x16 f16): lane holds column N = t*16 + m16,
  // halves i=0..15 -> K = kc*32 + hi*16 + i  (16 consecutive values, row N).
  v16h B2f[2][4];
#pragma unroll
  for (int t = 0; t < 4; ++t) {
#pragma unroll
    for (int kc = 0; kc < 2; ++kc) {
      B2f[kc][t] = cvt_frag16(W2 + (t * 16 + m16) * COUT + kc * 32 + hi * 16);
    }
  }

  const int nWaves = gridDim.x * 8;
  for (int pt = blockIdx.x * 8 + wv; pt < BATCH * NPTS; pt += nWaves) {
    const int b = pt >> 12;
    const int n = pt & (NPTS - 1);
    const float* xb = xyz + (size_t)b * NPTS * 3;

    // ---- gather 16 neighbors, relative coords (lanes 0..15 own one row) --
    const float cx = xb[n * 3 + 0], cy = xb[n * 3 + 1], cz = xb[n * 3 + 2];
    float gx = 0.f, gy = 0.f, gz = 0.f;
    if (lane < KNN) {
      const int p = idx[((size_t)pt << 4) + lane];
      gx = xb[p * 3 + 0] - cx;
      gy = xb[p * 3 + 1] - cy;
      gz = xb[p * 3 + 2] - cz;
    }
    const float bx = __shfl(gx, m16);
    const float by = __shfl(gy, m16);
    const float bz = __shfl(gz, m16);

    // ---- layer 1: 3 -> 64, relu, straight into A fragments --------------
    // A (16x32 f16): lane row M = m16; half i -> K = hi*16 + i
    v16h A0, A1;
#pragma unroll
    for (int i = 0; i < 16; ++i) {
      const int o0 = hi * 16 + i;   // K block 0..31
      const int o1 = o0 + 32;                              // K block 32..63
      float a = fmaf(bx, W1s[o0 * 3 + 0], fmaf(by, W1s[o0 * 3 + 1], bz * W1s[o0 * 3 + 2]));
      float c = fmaf(bx, W1s[o1 * 3 + 0], fmaf(by, W1s[o1 * 3 + 1], bz * W1s[o1 * 3 + 2]));
      A0[i] = (_Float16)fmaxf(a, 0.f);
      A1[i] = (_Float16)fmaxf(c, 0.f);
    }

    // ---- layer 2: 64 -> 64 via WMMA, relu, into LDS (row-major 16x64) ----
#pragma unroll
    for (int t = 0; t < 4; ++t) {
      v8f c = {};
      c = __builtin_amdgcn_wmma_f32_16x16x32_f16(false, A0, false, B2f[0][t],
                                                 (short)0, c, false, false);
      c = __builtin_amdgcn_wmma_f32_16x16x32_f16(false, A1, false, B2f[1][t],
                                                 (short)0, c, false, false);
#pragma unroll
      for (int r = 0; r < 8; ++r) {
        H[(r + hi * 8) * COUT + t * 16 + m16] = (_Float16)fmaxf(c[r], 0.f);
      }
    }

    // ---- layer 3 A fragments from LDS (per-wave in-order, no barrier) ----
    v16h A20, A21;
    {
      const _Float16* r0 = H + m16 * COUT + hi * 16;
      A20 = combine8(*(const v8h*)(r0),      *(const v8h*)(r0 + 8));
      A21 = combine8(*(const v8h*)(r0 + 32), *(const v8h*)(r0 + 40));
    }

    // ---- layer 3: 64 -> 64 via WMMA, max over 16 neighbor rows, store ----
#pragma unroll
    for (int t = 0; t < 4; ++t) {
      const _Float16* w = W3h + (t * 16 + m16) * COUT + hi * 16;
      v16h B0 = combine8(*(const v8h*)(w),      *(const v8h*)(w + 8));
      v16h B1 = combine8(*(const v8h*)(w + 32), *(const v8h*)(w + 40));
      v8f c = {};
      c = __builtin_amdgcn_wmma_f32_16x16x32_f16(false, A20, false, B0,
                                                 (short)0, c, false, false);
      c = __builtin_amdgcn_wmma_f32_16x16x32_f16(false, A21, false, B1,
                                                 (short)0, c, false, false);
      float m = c[0];
#pragma unroll
      for (int r = 1; r < 8; ++r) m = fmaxf(m, c[r]);
      m = fmaxf(m, __shfl_xor(m, 16));        // combine the two M halves
      if (lane < 16) {
        // out layout: (B, 64, N)
        out[((size_t)(b * COUT + t * 16 + lane) << 12) + n] = m;
      }
    }
  }
}

// --------------------------------------------------------------------------
extern "C" void kernel_launch(void* const* d_in, const int* in_sizes, int n_in,
                              void* d_out, int out_size, void* d_ws, size_t ws_size,
                              hipStream_t stream) {
  const float* xyz = (const float*)d_in[0];
  const float* W1  = (const float*)d_in[1];
  const float* W2  = (const float*)d_in[2];
  const float* W3  = (const float*)d_in[3];
  float* out = (float*)d_out;
  int* idx = (int*)d_ws;  // B*N*16 ints = 2 MiB of workspace

  knn_kernel<<<BATCH * (NPTS / 256), 256, 0, stream>>>(xyz, idx);
  // 512 blocks * 8 waves = 4096 waves; 8 points per wave amortizes the
  // register-resident W2 fragments.
  mlp_kernel<<<512, 256, 0, stream>>>(xyz, W1, W2, W3, idx, out);
}
